// LSTMContinuousActorCritic_10007273800444
// MI455X (gfx1250) — matrix operand order repaired and checked by reading, not compiled, not run on hardware
//
#include <hip/hip_runtime.h>
#include <cstdint>
#include <cstddef>

// ---------------------------------------------------------------------------
// Types for WMMA fragments (gfx1250, wave32)
// ---------------------------------------------------------------------------
typedef __attribute__((ext_vector_type(16))) __bf16          v16bf;
typedef __attribute__((ext_vector_type(8)))  float           v8f;
typedef __attribute__((ext_vector_type(8)))  unsigned short  us8;
typedef __attribute__((ext_vector_type(16))) unsigned short  us16;

union BFrag {
    us16  us;
    us8   h[2];
    v16bf bf;
};

__device__ __forceinline__ unsigned short f2bf(float f) {
    union { float f; unsigned int u; } v; v.f = f;
    unsigned int u = v.u;
    u += 0x7FFFu + ((u >> 16) & 1u);          // round-to-nearest-even
    return (unsigned short)(u >> 16);
}
__device__ __forceinline__ float sigf(float x) { return 1.0f / (1.0f + __expf(-x)); }

// Problem constants
#define BB   256
#define TT   128
#define OBSN 256
#define HH   1024
#define AA   16
#define BT   (BB * TT)        // 32768

// d_out offsets (floats), in reference return order
#define OFF_ACT 0LL
#define OFF_LP  524288LL
#define OFF_ENT 557056LL
#define OFF_VAL 589824LL
#define OFF_HT  622592LL
#define OFF_CT  1146880LL

// ---------------------------------------------------------------------------
// Weight packing: f32 [4096,K1] + [4096,K2] -> bf16 [4096, K1+K2]
// ---------------------------------------------------------------------------
__global__ void pack2_bf16(const float* __restrict__ s1, int K1,
                           const float* __restrict__ s2, int K2,
                           unsigned short* __restrict__ dst) {
    const int row = blockIdx.x;
    const int K = K1 + K2;
    for (int c = threadIdx.x; c < K; c += blockDim.x) {
        float v = (c < K1) ? s1[(size_t)row * K1 + c]
                           : s2[(size_t)row * K2 + (c - K1)];
        dst[(size_t)row * K + c] = f2bf(v);
    }
}

// Head weights: rows 0..15 = Wa, row 16 = Wc, rows 17..31 = 0
__global__ void pack_head_bf16(const float* __restrict__ Wa,
                               const float* __restrict__ Wc,
                               unsigned short* __restrict__ dst) {
    const int row = blockIdx.x;   // 0..31
    for (int c = threadIdx.x; c < HH; c += blockDim.x) {
        float v = (row < 16) ? Wa[(size_t)row * HH + c]
                             : ((row == 16) ? Wc[c] : 0.0f);
        dst[(size_t)row * HH + c] = f2bf(v);
    }
}

// ---------------------------------------------------------------------------
// State init from h0/c0 inputs ([2,256,1024] each)
// ---------------------------------------------------------------------------
__global__ void init_state(const float* __restrict__ h0in,
                           const float* __restrict__ c0in,
                           float* __restrict__ h0f, float* __restrict__ h1f,
                           float* __restrict__ c0f, float* __restrict__ c1f) {
    const int idx = blockIdx.x * blockDim.x + threadIdx.x;   // 0..262143
    h0f[idx] = h0in[idx];
    h1f[idx] = h0in[262144 + idx];
    c0f[idx] = c0in[idx];
    c1f[idx] = c0in[262144 + idx];
}

// ---------------------------------------------------------------------------
// Per-step prep: apply done mask to state, stage x_t and h_prev as bf16
// A0 layout [256,1280]: cols 0..255 = x_t, 256..1279 = h0_prev(masked)
// A1 layout [256,2048]: cols 0..1023 = h0_new (gemm0 writes), 1024..2047 = h1_prev(masked)
// ---------------------------------------------------------------------------
__global__ void prep_step(const float* __restrict__ x,
                          const int* __restrict__ done, int t,
                          float* __restrict__ h0f, float* __restrict__ h1f,
                          float* __restrict__ c0f, float* __restrict__ c1f,
                          unsigned short* __restrict__ A0,
                          unsigned short* __restrict__ A1) {
    const int b   = blockIdx.x;
    const int tid = threadIdx.x;
    const float m = 1.0f - (float)done[b * TT + t];
    // stage x_t -> bf16
    const float xv = x[((size_t)b * TT + t) * OBSN + tid];
    A0[(size_t)b * 1280 + tid] = f2bf(xv);
#pragma unroll
    for (int it = 0; it < 4; ++it) {
        const int j = tid + it * 256;
        const size_t idx = (size_t)b * HH + j;
        float h0v = h0f[idx] * m; h0f[idx] = h0v;
        A0[(size_t)b * 1280 + 256 + j] = f2bf(h0v);
        float h1v = h1f[idx] * m; h1f[idx] = h1v;
        A1[(size_t)b * 2048 + 1024 + j] = f2bf(h1v);
        c0f[idx] *= m;
        c1f[idx] *= m;
    }
}

// ---------------------------------------------------------------------------
// Fused GEMM + LSTM cell.
//   z = Acat @ Wcat^T + bias   (4 gates), then c' = sig(f)*c + sig(i)*tanh(g),
//   h' = sig(o)*tanh(c').
// Tiling: block = 8 waves; wave w handles j-tile (blockIdx.x*8+w) for ALL 4
// gates over a 16-row batch tile (blockIdx.y). Grid = (8, 16) -> 1024 waves.
// The 16-row A tile is shared by all 8 waves: staged ONCE into LDS with
// global_load_async_to_lds_b128 (ASYNCcnt), then consumed via ds_load_b128.
// hbout: bf16 copy of h' (A1 for layer0; hidden trace for layer1).
// ---------------------------------------------------------------------------
template <int KTOT>
__global__ void __launch_bounds__(256, 1)
lstm_gemm_cell(const unsigned short* __restrict__ Abf,
               const unsigned short* __restrict__ Wbf,
               const float* __restrict__ bias,
               float* __restrict__ cbuf,
               float* __restrict__ hfout,
               unsigned short* __restrict__ hbout,
               long long hb_stride) {
    __shared__ __align__(16) unsigned short Atile[16 * KTOT];

    const int lane = threadIdx.x & 31;
    const int wave = threadIdx.x >> 5;
    const int half = lane >> 4;        // K-half select (ISA A/B lane split)
    const int l16  = lane & 15;
    const int jt   = blockIdx.x * 8 + wave;   // 0..63
    const int bt   = blockIdx.y;              // 0..15

    // ---- Stage A tile (16 rows x KTOT bf16) global -> LDS, async DMA ----
    {
        constexpr int CHUNKS = (16 * KTOT * 2) / 16;   // 16B chunks
        const char* gbase = (const char*)(Abf + (size_t)bt * 16 * KTOT);
        const unsigned lds_base = (unsigned)(unsigned long long)(&Atile[0]);
        for (int c = threadIdx.x; c < CHUNKS; c += blockDim.x) {
            unsigned lds_addr = lds_base + (unsigned)(c * 16);
            const char* gaddr = gbase + (size_t)c * 16;
            asm volatile("global_load_async_to_lds_b128 %0, %1, off"
                         :
                         : "v"(lds_addr), "v"(gaddr)
                         : "memory");
        }
        asm volatile("s_wait_asynccnt 0x0" ::: "memory");
        __syncthreads();
    }

    v8f ai = {}, af = {}, ag = {}, ao = {};

    const unsigned short* arow_l = &Atile[(size_t)l16 * KTOT];
    const size_t gstep = (size_t)HH * KTOT;
    const unsigned short* __restrict__ wrow =
        Wbf + (size_t)(jt * 16 + l16) * KTOT + half * 16;

#pragma unroll 2
    for (int k0 = 0; k0 < KTOT; k0 += 32) {
        BFrag a;
        a.h[0] = *(const us8*)(arow_l + k0 + half * 16);       // ds_load_b128
        a.h[1] = *(const us8*)(arow_l + k0 + half * 16 + 8);

        BFrag w;
        w.us = *(const us16*)(wrow + k0);
        ai = __builtin_amdgcn_wmma_f32_16x16x32_bf16(false, a.bf, false, w.bf,
                                                     (short)0, ai, false, false);
        w.us = *(const us16*)(wrow + gstep + k0);
        af = __builtin_amdgcn_wmma_f32_16x16x32_bf16(false, a.bf, false, w.bf,
                                                     (short)0, af, false, false);
        w.us = *(const us16*)(wrow + 2 * gstep + k0);
        ag = __builtin_amdgcn_wmma_f32_16x16x32_bf16(false, a.bf, false, w.bf,
                                                     (short)0, ag, false, false);
        w.us = *(const us16*)(wrow + 3 * gstep + k0);
        ao = __builtin_amdgcn_wmma_f32_16x16x32_bf16(false, a.bf, false, w.bf,
                                                     (short)0, ao, false, false);
        // pull next K block of weights toward the WGP (global_prefetch_b8)
        __builtin_prefetch(wrow + k0 + 32, 0, 1);
    }

    // Epilogue: fused LSTM cell on the 16x16 D tiles.
    // D layout: VGPR r -> M = r + half*8 (lanes 0-15 / 16-31), N = l16.
    const int j = jt * 16 + l16;
    const float bi_ = bias[j];
    const float bf_ = bias[HH + j];
    const float bg_ = bias[2 * HH + j];
    const float bo_ = bias[3 * HH + j];
#pragma unroll
    for (int r = 0; r < 8; ++r) {
        const int m = bt * 16 + half * 8 + r;         // batch row
        const size_t idx = (size_t)m * HH + j;
        const float zi = ai[r] + bi_;
        const float zf = af[r] + bf_;
        const float zg = ag[r] + bg_;
        const float zo = ao[r] + bo_;
        const float cold = cbuf[idx];
        const float cn = sigf(zf) * cold + sigf(zi) * tanhf(zg);
        const float hn = sigf(zo) * tanhf(cn);
        cbuf[idx]  = cn;
        hfout[idx] = hn;
        hbout[(size_t)m * hb_stride + j] = f2bf(hn);
    }
}

// ---------------------------------------------------------------------------
// Head GEMM: hidden[32768,1024] @ Whead^T[1024,32] -> meanval[32768,32]
// (cols 0..15 = mean + ba, col 16 = value + bc). Each wave owns its own row
// tile, so no cross-wave A reuse -> keep direct global loads.
// ---------------------------------------------------------------------------
__global__ void __launch_bounds__(256, 1)
head_gemm(const unsigned short* __restrict__ hidden,
          const unsigned short* __restrict__ Whead,
          const float* __restrict__ ba, const float* __restrict__ bc,
          float* __restrict__ meanval) {
    const int lane = threadIdx.x & 31;
    const int wave = threadIdx.x >> 5;
    const int half = lane >> 4;
    const int l16  = lane & 15;
    const int rt   = blockIdx.x * 8 + wave;   // 0..2047 row tiles

    v8f acc0 = {}, acc1 = {};
    const unsigned short* __restrict__ arow =
        hidden + (size_t)(rt * 16 + l16) * HH;
    const unsigned short* __restrict__ w0 =
        Whead + (size_t)l16 * HH + half * 16;
    const unsigned short* __restrict__ w1 =
        Whead + (size_t)(16 + l16) * HH + half * 16;

#pragma unroll 2
    for (int k0 = 0; k0 < HH; k0 += 32) {
        BFrag a;
        a.h[0] = *(const us8*)(arow + k0 + half * 16);
        a.h[1] = *(const us8*)(arow + k0 + half * 16 + 8);
        BFrag w;
        w.us = *(const us16*)(w0 + k0);
        acc0 = __builtin_amdgcn_wmma_f32_16x16x32_bf16(false, a.bf, false, w.bf,
                                                       (short)0, acc0, false, false);
        w.us = *(const us16*)(w1 + k0);
        acc1 = __builtin_amdgcn_wmma_f32_16x16x32_bf16(false, a.bf, false, w.bf,
                                                       (short)0, acc1, false, false);
        __builtin_prefetch(arow + k0 + 32, 0, 1);
    }

    const float add0 = ba[l16];
    const float add1 = (l16 == 0) ? bc[0] : 0.0f;
#pragma unroll
    for (int r = 0; r < 8; ++r) {
        const int m = rt * 16 + half * 8 + r;
        meanval[(size_t)m * 32 + l16]      = acc0[r] + add0;
        meanval[(size_t)m * 32 + 16 + l16] = acc1[r] + add1;
    }
}

// ---------------------------------------------------------------------------
// Final elementwise: tanh-Gaussian log-prob, entropy, value, action copy
// ---------------------------------------------------------------------------
__global__ void heads_final(const float* __restrict__ action,
                            const float* __restrict__ meanval,
                            const float* __restrict__ logstd,
                            float* __restrict__ out) {
    const int row = blockIdx.x * blockDim.x + threadIdx.x;   // 0..32767
    const float LOG2PI = 1.8378770664093453f;
    const float LN2    = 0.6931471805599453f;
    float lp = 0.0f, corr = 0.0f, ent = 0.0f;
#pragma unroll
    for (int a = 0; a < AA; ++a) {
        const float act = action[(size_t)row * AA + a];
        out[OFF_ACT + (size_t)row * AA + a] = act;
        float y = fminf(fmaxf(act, -0.999f), 0.999f);
        y = fminf(fmaxf(y, -1.0f + 1e-6f), 1.0f - 1e-6f);
        const float raw = 0.5f * __logf((1.0f + y) / (1.0f - y));
        const float ls  = logstd[a];
        const float inv_std = __expf(-ls);
        const float mean = meanval[(size_t)row * 32 + a];
        const float d = (raw - mean) * inv_std;
        lp   += -0.5f * d * d - ls - 0.5f * LOG2PI;
        corr += 2.0f * (LN2 - raw - log1pf(__expf(-2.0f * raw)));
        ent  += 0.5f * (1.0f + LOG2PI) + ls;
    }
    out[OFF_LP  + row] = lp - corr;
    out[OFF_ENT + row] = ent;
    out[OFF_VAL + row] = meanval[(size_t)row * 32 + 16];
}

// ---------------------------------------------------------------------------
// hT / cT copy-out
// ---------------------------------------------------------------------------
__global__ void copy_state_out(const float* __restrict__ h0f,
                               const float* __restrict__ h1f,
                               const float* __restrict__ c0f,
                               const float* __restrict__ c1f,
                               float* __restrict__ out) {
    const int idx = blockIdx.x * blockDim.x + threadIdx.x;   // 0..262143
    out[OFF_HT + idx]          = h0f[idx];
    out[OFF_HT + 262144 + idx] = h1f[idx];
    out[OFF_CT + idx]          = c0f[idx];
    out[OFF_CT + 262144 + idx] = c1f[idx];
}

// ---------------------------------------------------------------------------
// Host launch
// ---------------------------------------------------------------------------
extern "C" void kernel_launch(void* const* d_in, const int* in_sizes, int n_in,
                              void* d_out, int out_size, void* d_ws, size_t ws_size,
                              hipStream_t stream) {
    (void)in_sizes; (void)n_in; (void)out_size; (void)ws_size;
    // Inputs per setup_inputs() order
    const float* x      = (const float*)d_in[0];
    const float* h0in   = (const float*)d_in[1];
    const float* c0in   = (const float*)d_in[2];
    const int*   done   = (const int*)  d_in[3];
    const float* action = (const float*)d_in[4];
    const float* W_ih0  = (const float*)d_in[5];
    const float* W_hh0  = (const float*)d_in[6];
    const float* b0     = (const float*)d_in[7];
    const float* W_ih1  = (const float*)d_in[8];
    const float* W_hh1  = (const float*)d_in[9];
    const float* b1     = (const float*)d_in[10];
    const float* Wa     = (const float*)d_in[11];
    const float* ba     = (const float*)d_in[12];
    const float* logstd = (const float*)d_in[13];
    const float* Wc     = (const float*)d_in[14];
    const float* bc     = (const float*)d_in[15];
    float* out = (float*)d_out;

    // Workspace carve (256B aligned)
    char* wp = (char*)d_ws;
    auto carve = [&](size_t bytes) -> void* {
        void* p = (void*)wp;
        wp += (bytes + 255) & ~(size_t)255;
        return p;
    };
    unsigned short* Wcat0  = (unsigned short*)carve((size_t)4096 * 1280 * 2); // 10.5 MB
    unsigned short* Wcat1  = (unsigned short*)carve((size_t)4096 * 2048 * 2); // 16.8 MB
    unsigned short* Whead  = (unsigned short*)carve((size_t)32 * HH * 2);
    unsigned short* A0     = (unsigned short*)carve((size_t)BB * 1280 * 2);
    unsigned short* A1     = (unsigned short*)carve((size_t)BB * 2048 * 2);
    float* h0f = (float*)carve((size_t)BB * HH * 4);
    float* h1f = (float*)carve((size_t)BB * HH * 4);
    float* c0f = (float*)carve((size_t)BB * HH * 4);
    float* c1f = (float*)carve((size_t)BB * HH * 4);
    unsigned short* hidden = (unsigned short*)carve((size_t)BT * HH * 2);     // 64 MB
    float* meanval = (float*)carve((size_t)BT * 32 * 4);                      // 4 MB

    // 1) Pack weights to bf16 (resident in 192MB L2 across the scan)
    pack2_bf16<<<4096, 256, 0, stream>>>(W_ih0, OBSN, W_hh0, HH, Wcat0);
    pack2_bf16<<<4096, 256, 0, stream>>>(W_ih1, HH,   W_hh1, HH, Wcat1);
    pack_head_bf16<<<32, 256, 0, stream>>>(Wa, Wc, Whead);

    // 2) Init recurrent state from inputs
    init_state<<<1024, 256, 0, stream>>>(h0in, c0in, h0f, h1f, c0f, c1f);

    // 3) Sequential scan: 128 x (prep, gemm0+cell, gemm1+cell)
    const dim3 ggrid(8, 16);
    for (int t = 0; t < TT; ++t) {
        prep_step<<<BB, 256, 0, stream>>>(x, done, t, h0f, h1f, c0f, c1f, A0, A1);
        // layer0: h0' bf16 lands in A1 cols 0..1023 (row stride 2048)
        lstm_gemm_cell<1280><<<ggrid, 256, 0, stream>>>(
            A0, Wcat0, b0, c0f, h0f, A1, 2048LL);
        // layer1: h1' bf16 lands in hidden[(b*T + t), :] (row stride T*H)
        lstm_gemm_cell<2048><<<ggrid, 256, 0, stream>>>(
            A1, Wcat1, b1, c1f, h1f, hidden + (size_t)t * HH, (long long)TT * HH);
    }

    // 4) Policy/value heads over the full hidden trace
    head_gemm<<<256, 256, 0, stream>>>(hidden, Whead, ba, bc, meanval);

    // 5) Log-prob / entropy / value / action copy
    heads_final<<<BT / 256, 256, 0, stream>>>(action, meanval, logstd, out);

    // 6) hT / cT
    copy_state_out<<<1024, 256, 0, stream>>>(h0f, h1f, c0f, c1f, out);
}
